// PAConv_offset_69861938036984
// MI455X (gfx1250) — matrix operand order repaired and checked by reading, not compiled, not run on hardware
//
#include <hip/hip_runtime.h>

// ---------------------------------------------------------------------------
// PAConv pipeline for MI455X (gfx1250, wave32, WMMA + TDM).
// All GEMM-like work (oc2 128->64, dcn 64->64, both 3x3 convs as implicit
// GEMMs) runs on v_wmma_f32_16x16x32_f16 with f32 accumulation. x is staged
// once as NHWC f16 (67MB -> resident in 192MB L2) so the 9-tap conv re-reads
// and the bilinear gathers are L2 hits. The oc2/dcn weight fragments (24KB)
// are DMA'd into LDS once per block via tensor_load_to_lds (TDM), removing
// ~786MB of per-tile L2 fragment re-reads. ~90 GFLOP total => memory-bound
// at f16 matrix rates; target is the ~30us HBM roofline.
// ---------------------------------------------------------------------------

#define HH 256
#define WW 256
#define NB 8
#define NFC 64
#define HWSZ (HH * WW)        // 65536
#define PIX (NB * HWSZ)       // 524288 pixels

typedef __attribute__((ext_vector_type(16))) _Float16 v16h;
typedef __attribute__((ext_vector_type(8)))  float    v8f;
typedef __attribute__((ext_vector_type(4))) unsigned uv4;
typedef __attribute__((ext_vector_type(8)))  int      iv8;
typedef __attribute__((ext_vector_type(4)))  int      iv4;

union Frag16 { v16h h; _Float16 f[16]; uv4 q[2]; };
union H8     { uv4 q; _Float16 f[8]; };

// D = A(16x32 f16) x B(32x16 f16) + C(16x16 f32)
__device__ __forceinline__ v8f wmma_f16(v16h a, v16h b, v8f c) {
  return __builtin_amdgcn_wmma_f32_16x16x32_f16(
      /*neg_a=*/false, a, /*neg_b=*/false, b,
      /*c_mod=*/(short)0, c, /*reuse_a=*/false, /*reuse_b=*/false);
}

// B fragments stored pre-packed: frag*512 + lane*16 + e, contiguous per lane.
// Works for both global (global_load_b128) and LDS (ds_load_b128) sources.
__device__ __forceinline__ v16h load_bfrag(const _Float16* wb, int frag,
                                           int lane) {
  Frag16 f;
  const uv4* s = (const uv4*)(wb + frag * 512 + lane * 16);
  f.q[0] = s[0];
  f.q[1] = s[1];
  return f.h;
}

// ---------------------------------------------------------------------------
// Kernel 0a: x NCHW f32 -> NHWC f16 (LDS transpose for coalescing both sides)
// ---------------------------------------------------------------------------
__global__ void k_x_to_nhwc(const float* __restrict__ x,
                            _Float16* __restrict__ xh) {
  __shared__ unsigned st[8][32][33];   // [wave][pixel][ch-pair], padded
  int wave = threadIdx.x >> 5, lane = threadIdx.x & 31;
  int p0 = (blockIdx.x * 8 + wave) * 32;       // 32 pixels per wave, same batch
  int b  = p0 >> 16;
  int hw = (p0 & (HWSZ - 1)) + lane;
  const float* xb = x + (long)b * NFC * HWSZ + hw;
#pragma unroll
  for (int c = 0; c < NFC; c += 2) {
    union { _Float16 h[2]; unsigned u; } pk;
    pk.h[0] = (_Float16)xb[(long)c * HWSZ];
    pk.h[1] = (_Float16)xb[(long)(c + 1) * HWSZ];
    st[wave][lane][c >> 1] = pk.u;
  }
  __syncthreads();
  unsigned* outp = (unsigned*)xh + (long)p0 * 32;   // 32 u32 = 64 f16 / pixel
#pragma unroll
  for (int i = 0; i < 32; ++i)
    outp[i * 32 + lane] = st[wave][i][lane];
}

// ---------------------------------------------------------------------------
// Kernel 0b: pack all GEMM weights into WMMA B-fragment layout (f16).
// B elem e of lane L: N = L&15, K = (e&7) + 8*(L>>4) + 16*(e>>3)  (dense 32x16 f16 B).
// wb2 and wbd MUST be contiguous in the workspace (single TDM tile).
// ---------------------------------------------------------------------------
__global__ void k_pack_weights(const float* __restrict__ oc2w,
                               const float* __restrict__ dcnw,
                               const float* __restrict__ k3w,
                               const float* __restrict__ k4w,
                               _Float16* __restrict__ wb2,
                               _Float16* __restrict__ wbd,
                               _Float16* __restrict__ wb3,
                               _Float16* __restrict__ wb4) {
  int idx = blockIdx.x * 256 + threadIdx.x;
  if (idx >= 168 * 512) return;
  int r = idx & 511, f = idx >> 9;
  int lane = r >> 4, e = r & 15;
  int n  = lane & 15;
  int kb = (e & 7) + 8 * (lane >> 4) + 16 * (e >> 3);
  if (f < 16) {                       // oc2: K=128 (kt 0..3), N=64 (nt 0..3)
    int kt = f >> 2, nt = f & 3;
    wb2[idx] = (_Float16)oc2w[(nt * 16 + n) * 128 + kt * 32 + kb];
  } else if (f < 24) {                // dcn: K=64 (kt 0..1)
    int g = f - 16; int kt = g >> 2, nt = g & 3;
    wbd[g * 512 + r] = (_Float16)dcnw[(nt * 16 + n) * 64 + kt * 32 + kb];
  } else if (f < 96) {                // k3: 9 taps x (kt 0..1) x (nt 0..3)
    int g = f - 24; int t = g >> 3, rem = g & 7, kt = rem >> 2, nt = rem & 3;
    wb3[g * 512 + r] = (_Float16)k3w[((nt * 16 + n) * 64 + kt * 32 + kb) * 9 + t];
  } else {                            // k4
    int g = f - 96; int t = g >> 3, rem = g & 7, kt = rem >> 2, nt = rem & 3;
    wb4[g * 512 + r] = (_Float16)k4w[((nt * 16 + n) * 64 + kt * 32 + kb) * 9 + t];
  }
}

// bilinear gather helper: accumulate w * xh[pixel, icb..icb+7]
__device__ __forceinline__ void gadd(const _Float16* __restrict__ xh, long pb,
                                     int icb, bool v, float w, float* a8) {
  if (v) {
    H8 t;
    t.q = *(const uv4*)(xh + pb + icb);
#pragma unroll
    for (int j = 0; j < 8; ++j) a8[j] = fmaf(w, (float)t.f[j], a8[j]);
  }
}

// ---------------------------------------------------------------------------
// Kernel 1: fused condition-MLP (oc1 scalar, oc2 WMMA) -> offsets/mask ->
//           bilinear sample of xh -> dcn (WMMA) -> sigmoid -> y (NHWC f16).
// One wave = one 16-pixel row tile; 8 waves / block; uniform control flow.
// oc2+dcn B-fragments (24KB contiguous) are staged into LDS once per block
// with a single Tensor Data Mover transfer.
// ---------------------------------------------------------------------------
__global__ void k_offset_dcn(const float* __restrict__ cond,
                             const float* __restrict__ oc1w,
                             const float* __restrict__ oc1b,
                             const float* __restrict__ oc2b,
                             const float* __restrict__ offw,
                             const float* __restrict__ offb,
                             const float* __restrict__ dcnb,
                             const _Float16* __restrict__ wb2, // wbd follows
                             const _Float16* __restrict__ xh,
                             _Float16* __restrict__ yws) {
  __shared__ float s_oc1w[128], s_oc1b[128], s_oc2b[64];
  __shared__ float s_offw[192], s_offb[3], s_dcnb[64];
  __shared__ float s_feat[8][16][66];   // [wave][pixel][ch], padded
  __shared__ float s_cw[8][16][5];      // {y0, x0, wy1, wx1, mask}
  __shared__ _Float16 s_wb[24 * 512];   // wb2 (16 frags) + wbd (8 frags)

  int tid = threadIdx.x;
  if (tid < 128) { s_oc1w[tid] = oc1w[tid]; s_oc1b[tid] = oc1b[tid]; }
  if (tid < 64)  { s_oc2b[tid] = oc2b[tid]; s_dcnb[tid] = dcnb[tid]; }
  if (tid < 192) s_offw[tid] = offw[tid];
  if (tid < 3)   s_offb[tid] = offb[tid];

#if __has_builtin(__builtin_amdgcn_tensor_load_to_lds) && \
    __has_builtin(__builtin_amdgcn_s_wait_tensorcnt)
  // --- TDM: one 24KB DMA (6144 dwords, 1 row) global -> LDS, wave 0 only ---
  if (tid == 0) {
    unsigned lds = (unsigned)(size_t)(void*)&s_wb[0];
    unsigned long long ga = (unsigned long long)(size_t)wb2;
    uv4 g0;
    g0[0] = 1u;                                   // count=1, user descriptor
    g0[1] = lds;                                  // lds_addr (bytes)
    g0[2] = (unsigned)(ga & 0xFFFFFFFFu);         // global_addr[31:0]
    g0[3] = (unsigned)((ga >> 32) & 0x1FFFFFFu)   // global_addr[56:32]
            | (2u << 30);                         // type=2 ("image")
    iv8 g1;
    g1[0] = (int)(2u << 16);          // workgroup_mask=0, data_size=4B
    g1[1] = (int)(0x1800u << 16);     // tensor_dim0[15:0]=6144 in bits[31:16]
    g1[2] = (int)(1u << 16);          // tensor_dim0 hi=0; tensor_dim1=1
    g1[3] = (int)(0x1800u << 16);     // tensor_dim1 hi=0; tile_dim0=6144
    g1[4] = 1;                        // tile_dim1=1, tile_dim2=0
    g1[5] = 6144;                     // tensor_dim0_stride (lo)
    g1[6] = 0;
    g1[7] = 0;
    iv4 gz4 = {0, 0, 0, 0};           // groups 2/3 unused (<=2D tensor)
    iv8 gz8 = {0, 0, 0, 0, 0, 0, 0, 0};
    // 6-arg form (clang-23 / therock-10.0 headers)
    __builtin_amdgcn_tensor_load_to_lds(g0, g1, gz4, gz4, gz8, 0);
    __builtin_amdgcn_s_wait_tensorcnt(0);
  }
#else
  // Fallback: cooperative vector copy global -> LDS.
  {
    const uv4* src = (const uv4*)wb2;
    uv4* dst = (uv4*)s_wb;
    for (int i = tid; i < 24 * 512 / 8; i += 256) dst[i] = src[i];
  }
#endif
  __syncthreads();

  int wave = tid >> 5, lane = tid & 31, m = lane & 15, hi = lane >> 4;
  int nloc = lane & 15;
  int tile = blockIdx.x * 8 + wave;
  int p0 = tile * 16;
  int b = p0 >> 16, hw = p0 & (HWSZ - 1), h = hw >> 8, w0 = hw & 255;

  float cv = cond[p0 + m];

  // oc1 (1->128) + lrelu, directly into A-fragment layout (16 px x 128 K).
  // A elem e of lane L: M = L&15, K = e + 8*((e>>3) + (L>>4)).
  Frag16 A1[4];
#pragma unroll
  for (int kt = 0; kt < 4; ++kt)
#pragma unroll
    for (int e = 0; e < 16; ++e) {
      int j = kt * 32 + e + 8 * ((e >> 3) + hi);
      float f1 = fmaf(s_oc1w[j], cv, s_oc1b[j]);
      A1[kt].f[e] = (_Float16)(f1 >= 0.f ? f1 : 0.2f * f1);
    }

  // oc2 (128->64) on the matrix pipe: 16 WMMAs, B from LDS (frags 0..15)
  v8f acc[4] = {};
#pragma unroll
  for (int nt = 0; nt < 4; ++nt)
#pragma unroll
    for (int kt = 0; kt < 4; ++kt)
      acc[nt] = wmma_f16(A1[kt].h, load_bfrag(s_wb, kt * 4 + nt, lane),
                         acc[nt]);

  // bias + lrelu, scatter into LDS in D layout (M = r + 8*hi, N = nloc)
#pragma unroll
  for (int nt = 0; nt < 4; ++nt) {
    float bb = s_oc2b[nt * 16 + nloc];
#pragma unroll
    for (int r = 0; r < 8; ++r) {
      float v = acc[nt][r] + bb;
      s_feat[wave][r + 8 * hi][nt * 16 + nloc] = (v >= 0.f ? v : 0.2f * v);
    }
  }
  __syncthreads();

  // 64->3 offset head: one lane per pixel
  if (hi == 0) {
    float o0 = s_offb[0], o1 = s_offb[1], o2 = s_offb[2];
#pragma unroll 8
    for (int ch = 0; ch < 64; ++ch) {
      float fv = s_feat[wave][m][ch];
      o0 = fmaf(s_offw[ch], fv, o0);
      o1 = fmaf(s_offw[64 + ch], fv, o1);
      o2 = fmaf(s_offw[128 + ch], fv, o2);
    }
    float msk = 1.f / (1.f + __expf(-o2));
    float py = (float)h + o0;
    float px = (float)(w0 + m) + o1;
    float y0f = floorf(py), x0f = floorf(px);
    s_cw[wave][m][0] = y0f;       s_cw[wave][m][1] = x0f;
    s_cw[wave][m][2] = py - y0f;  s_cw[wave][m][3] = px - x0f;
    s_cw[wave][m][4] = msk;
  }
  __syncthreads();

  // bilinear sample (4 taps, mask folded in) straight into dcn A-fragments
  int   y0  = (int)s_cw[wave][m][0], x0 = (int)s_cw[wave][m][1];
  float wy1 = s_cw[wave][m][2], wx1 = s_cw[wave][m][3], msk = s_cw[wave][m][4];
  float wy0 = 1.f - wy1, wx0 = 1.f - wx1;
  int y1 = y0 + 1, x1 = x0 + 1;
  bool vy0 = (y0 >= 0) & (y0 < HH), vy1 = (y1 >= 0) & (y1 < HH);
  bool vx0 = (x0 >= 0) & (x0 < WW), vx1 = (x1 >= 0) & (x1 < WW);
  float w00 = wy0 * wx0 * msk, w01 = wy0 * wx1 * msk;
  float w10 = wy1 * wx0 * msk, w11 = wy1 * wx1 * msk;
  long rowb = (long)b * HWSZ;
  long p00 = (rowb + y0 * WW + x0) * 64, p01 = (rowb + y0 * WW + x1) * 64;
  long p10 = (rowb + y1 * WW + x0) * 64, p11 = (rowb + y1 * WW + x1) * 64;

  Frag16 As[2];
#pragma unroll
  for (int kt = 0; kt < 2; ++kt)
#pragma unroll
    for (int g = 0; g < 2; ++g) {
      int icb = kt * 32 + g * 16 + 8 * hi;   // 8 contiguous ic per group
      float a8[8] = {};
      gadd(xh, p00, icb, vy0 && vx0, w00, a8);
      gadd(xh, p01, icb, vy0 && vx1, w01, a8);
      gadd(xh, p10, icb, vy1 && vx0, w10, a8);
      gadd(xh, p11, icb, vy1 && vx1, w11, a8);
#pragma unroll
      for (int j = 0; j < 8; ++j) As[kt].f[g * 8 + j] = (_Float16)a8[j];
    }

  // dcn 64->64 (8 WMMAs, B from LDS frags 16..23) + bias + sigmoid -> y
  v8f acc2[4] = {};
#pragma unroll
  for (int nt = 0; nt < 4; ++nt)
#pragma unroll
    for (int kt = 0; kt < 2; ++kt)
      acc2[nt] = wmma_f16(As[kt].h, load_bfrag(s_wb, 16 + kt * 4 + nt, lane),
                          acc2[nt]);
#pragma unroll
  for (int nt = 0; nt < 4; ++nt) {
    float bb = s_dcnb[nt * 16 + nloc];
#pragma unroll
    for (int r = 0; r < 8; ++r) {
      float v = acc2[nt][r] + bb;
      v = 1.f / (1.f + __expf(-v));
      yws[(long)(p0 + r + 8 * hi) * 64 + nt * 16 + nloc] = (_Float16)v;
    }
  }
}

// ---------------------------------------------------------------------------
// Kernels 2/3: 3x3 conv as 9-tap implicit GEMM, 72 WMMAs per 16-pixel tile.
// MODE 0: out1 = conv(xh, k3) * y -> tws (NHWC f16)
// MODE 1: out  = conv(tws, k4)    -> d_out (NCHW f32)
// B-fragments (72KB) stay in L2 (won't fit in 64KB static LDS); each frag is
// amortized over a full 16x16x32 WMMA.
// ---------------------------------------------------------------------------
template <int MODE>
__global__ void k_conv3x3(const _Float16* __restrict__ in,
                          const _Float16* __restrict__ wb,
                          const _Float16* __restrict__ yws,
                          _Float16* __restrict__ out16,
                          float* __restrict__ out32) {
  int tid = threadIdx.x;
  int wave = tid >> 5, lane = tid & 31, m = lane & 15, hi = lane >> 4;
  int nloc = lane & 15;
  int tile = blockIdx.x * 8 + wave;
  int p0 = tile * 16;
  int b = p0 >> 16, hw = p0 & (HWSZ - 1), h = hw >> 8, w0 = hw & 255;
  int w = w0 + m;

  __builtin_prefetch(wb + lane * 16, 0, 3);   // weights live in L2

  v8f acc[4] = {};
#pragma unroll
  for (int t = 0; t < 9; ++t) {
    int ky = t / 3 - 1, kx = t % 3 - 1;
    int yy = h + ky, xx = w + kx;
    bool valid = (yy >= 0) & (yy < HH) & (xx >= 0) & (xx < WW);
    long pb = ((long)b * HWSZ + yy * WW + xx) * 64;
#pragma unroll
    for (int kt = 0; kt < 2; ++kt) {
      Frag16 a;
      if (valid) {
        const _Float16* ap = in + pb + kt * 32 + 8 * hi;
        a.q[0] = *(const uv4*)ap;          // K group 0: 8 contiguous f16
        a.q[1] = *(const uv4*)(ap + 16);   // K group 1
      } else {
        uv4 z = {0, 0, 0, 0};
        a.q[0] = z; a.q[1] = z;            // zero padding
      }
#pragma unroll
      for (int nt = 0; nt < 4; ++nt)
        acc[nt] = wmma_f16(a.h, load_bfrag(wb, (t * 2 + kt) * 4 + nt, lane),
                           acc[nt]);
    }
  }

#pragma unroll
  for (int nt = 0; nt < 4; ++nt)
#pragma unroll
    for (int r = 0; r < 8; ++r) {
      int mm = r + 8 * hi, ch = nt * 16 + nloc;
      if (MODE == 0) {
        float yv = (float)yws[(long)(p0 + mm) * 64 + ch];
        out16[(long)(p0 + mm) * 64 + ch] = (_Float16)(acc[nt][r] * yv);
      } else {
        out32[(long)(b * NFC + ch) * HWSZ + h * WW + (w0 + mm)] = acc[nt][r];
      }
    }
}

// ---------------------------------------------------------------------------
extern "C" void kernel_launch(void* const* d_in, const int* in_sizes, int n_in,
                              void* d_out, int out_size, void* d_ws,
                              size_t ws_size, hipStream_t stream) {
  const float* x    = (const float*)d_in[0];
  const float* cond = (const float*)d_in[1];
  const float* oc1w = (const float*)d_in[2];
  const float* oc1b = (const float*)d_in[3];
  const float* oc2w = (const float*)d_in[4];
  const float* oc2b = (const float*)d_in[5];
  const float* offw = (const float*)d_in[6];
  const float* offb = (const float*)d_in[7];
  const float* dcnw = (const float*)d_in[8];
  const float* dcnb = (const float*)d_in[9];
  const float* k3w  = (const float*)d_in[10];
  const float* k4w  = (const float*)d_in[11];
  float* out = (float*)d_out;

  char* p = (char*)d_ws;
  _Float16* xh  = (_Float16*)p; p += (size_t)PIX * 64 * 2;   // 67 MB NHWC f16
  _Float16* yws = (_Float16*)p; p += (size_t)PIX * 64 * 2;   // gate y
  _Float16* tws = (_Float16*)p; p += (size_t)PIX * 64 * 2;   // conv3 * y
  _Float16* wb2 = (_Float16*)p; p += 16 * 512 * 2;           // | contiguous
  _Float16* wbd = (_Float16*)p; p += 8 * 512 * 2;            // | 24KB TDM tile
  _Float16* wb3 = (_Float16*)p; p += 72 * 512 * 2;
  _Float16* wb4 = (_Float16*)p; p += 72 * 512 * 2;

  const int tileBlocks = PIX / 16 / 8;   // 4096 blocks, 8 waves each

  k_x_to_nhwc<<<PIX / 256, 256, 0, stream>>>(x, xh);
  k_pack_weights<<<(168 * 512 + 255) / 256, 256, 0, stream>>>(
      oc2w, dcnw, k3w, k4w, wb2, wbd, wb3, wb4);
  k_offset_dcn<<<tileBlocks, 256, 0, stream>>>(
      cond, oc1w, oc1b, oc2b, offw, offb, dcnb, wb2, xh, yws);
  k_conv3x3<0><<<tileBlocks, 256, 0, stream>>>(xh, wb3, yws, tws, nullptr);
  k_conv3x3<1><<<tileBlocks, 256, 0, stream>>>(tws, wb4, nullptr, nullptr, out);
}
